// CQAttention_20718922236643
// MI455X (gfx1250) — matrix operand order repaired and checked by reading, not compiled, not run on hardware
//
#include <hip/hip_runtime.h>
#include <hip/hip_bf16.h>

#define LC 1024
#define LQ 512
#define BATCH 64
#define DD 256

typedef __bf16 bf16_t;
typedef __attribute__((ext_vector_type(16))) __bf16 v16bf;
typedef __attribute__((ext_vector_type(8)))  __bf16 v8bf;
typedef __attribute__((ext_vector_type(8)))  float  v8f;
typedef int v4i_ __attribute__((vector_size(16)));

// D = A(16x32 bf16) * B(32x16 bf16) + C(16x16 f32)
#define WMMA_BF16(A_, B_, C_) \
  __builtin_amdgcn_wmma_f32_16x16x32_bf16(false, (A_), false, (B_), (short)0, (C_), false, false)

// ---- Async 16B global->LDS copy (CDNA5 GLOBAL_LOAD_ASYNC_TO_LDS_B128), with fallback ----
__device__ inline void async_cp16(void* lds, const void* g) {
#if defined(__gfx1250__) && __has_builtin(__builtin_amdgcn_global_load_async_to_lds_b128)
  __builtin_amdgcn_global_load_async_to_lds_b128(
      (__attribute__((address_space(1))) v4i_*)(void*)g,
      (__attribute__((address_space(3))) v4i_*)lds, 0, 0);
#else
  *(v8bf*)lds = *(const v8bf*)g;
#endif
}
__device__ inline void async_wait() {
#if defined(__gfx1250__) && __has_builtin(__builtin_amdgcn_s_wait_asynccnt)
  __builtin_amdgcn_s_wait_asynccnt(0);
#elif defined(__AMDGCN__)
  asm volatile("s_wait_asynccnt 0" ::: "memory");
#endif
}

// ---- Fragment loaders from LDS tiles with row pitch 40 halves (80B, 16B-aligned) ----
// A (16x32, MxK): lanes 0-15 hold row M=lane, halves = K[0..7]+K[16..23];
//                 lanes 16-31 hold row M=lane-16, halves = K[8..15]+K[24..31].
__device__ inline v16bf frag_a(const bf16_t* base, int r, int hi) {
  union { v16bf v; v8bf h[2]; } u;
  const bf16_t* p = base + r * 40 + hi * 8;
  u.h[0] = *(const v8bf*)(p);
  u.h[1] = *(const v8bf*)(p + 16);
  return u.v;
}
// B (32x16, KxN) loaded from B^T stored row-major (N rows x 32 K):
// lanes 0-15: row N=lane, K[0..7]+K[16..23]; lanes 16-31: row N=lane-16, K[8..15]+K[24..31].
__device__ inline v16bf frag_b(const bf16_t* base, int r, int hi) {
  union { v16bf v; v8bf h[2]; } u;
  const bf16_t* p = base + r * 40 + hi * 8;
  u.h[0] = *(const v8bf*)(p);
  u.h[1] = *(const v8bf*)(p + 16);
  return u.v;
}

// ---- K0: out[b*Lrows + l] = dot(X[l,b,:], w) (+ bias) ----
__global__ __launch_bounds__(256) void rowdot_kernel(const float* __restrict__ X,
    const float* __restrict__ w, const float* __restrict__ bias, int use_bias,
    float* __restrict__ out, int Lrows) {
  int wave = threadIdx.x >> 5, lane = threadIdx.x & 31;
  int rid = blockIdx.x * 8 + wave;
  int l = rid / BATCH, b = rid % BATCH;
  const float* row = X + ((size_t)l * BATCH + b) * DD;
  float s = 0.f;
#pragma unroll
  for (int i = 0; i < DD / 32; ++i) s += row[lane + i * 32] * w[lane + i * 32];
#pragma unroll
  for (int off = 16; off >= 1; off >>= 1) s += __shfl_xor(s, off, 32);
  if (lane == 0) out[(size_t)b * Lrows + l] = s + (use_bias ? bias[0] : 0.f);
}

// ---- K1: S[b,c,q] = (C[c,b,:]*w4mlu) . Q[q,b,:] + sub0 + sub1  (split-bf16 WMMA) ----
__global__ __launch_bounds__(128) void s_kernel(const float* __restrict__ C,
    const float* __restrict__ Q, const float* __restrict__ w4mlu,
    const float* __restrict__ sub0, const float* __restrict__ sub1,
    float* __restrict__ S) {
  __shared__ bf16_t sChi[64 * 40], sClo[64 * 40], sQhi[64 * 40], sQlo[64 * 40];
  __shared__ float sDump[64 * 65];
  const int tid = threadIdx.x;
  const int b = blockIdx.z, c0 = blockIdx.y * 64, q0 = blockIdx.x * 64;
  const int wave = tid >> 5, lane = tid & 31;
  const int wc = wave >> 1, wq = wave & 1;
  const int row = lane & 15, hi = lane >> 4;
  const int sr = tid >> 1, scol = (tid & 1) * 16;
  v8f acc[2][2] = {};
  for (int k0 = 0; k0 < DD; k0 += 32) {
    const float* cp = C + ((size_t)(c0 + sr) * BATCH + b) * DD + k0 + scol;
    const float* qp = Q + ((size_t)(q0 + sr) * BATCH + b) * DD + k0 + scol;
    const float* wp = w4mlu + k0 + scol;
    if (k0 + 32 < DD) {           // prefetch next K-chunk (global_prefetch_b8)
      __builtin_prefetch(cp + 32, 0, 0);
      __builtin_prefetch(qp + 32, 0, 0);
    }
#pragma unroll
    for (int i = 0; i < 16; ++i) {
      float x = cp[i] * wp[i];
      bf16_t h = (bf16_t)x;
      sChi[sr * 40 + scol + i] = h;
      sClo[sr * 40 + scol + i] = (bf16_t)(x - (float)h);
      float y = qp[i];
      bf16_t g = (bf16_t)y;
      sQhi[sr * 40 + scol + i] = g;
      sQlo[sr * 40 + scol + i] = (bf16_t)(y - (float)g);
    }
    __syncthreads();
    v16bf ah[2], al[2], bh[2], bl[2];
#pragma unroll
    for (int i = 0; i < 2; ++i) {
      ah[i] = frag_a(sChi, wc * 32 + i * 16 + row, hi);
      al[i] = frag_a(sClo, wc * 32 + i * 16 + row, hi);
      bh[i] = frag_b(sQhi, wq * 32 + i * 16 + row, hi);
      bl[i] = frag_b(sQlo, wq * 32 + i * 16 + row, hi);
    }
#pragma unroll
    for (int i = 0; i < 2; ++i)
#pragma unroll
      for (int j = 0; j < 2; ++j) {
        acc[i][j] = WMMA_BF16(ah[i], bh[j], acc[i][j]);
        acc[i][j] = WMMA_BF16(ah[i], bl[j], acc[i][j]);
        acc[i][j] = WMMA_BF16(al[i], bh[j], acc[i][j]);
      }
    __syncthreads();
  }
#pragma unroll
  for (int i = 0; i < 2; ++i)
    for (int j = 0; j < 2; ++j)
#pragma unroll
      for (int r = 0; r < 8; ++r)
        sDump[(wc * 32 + i * 16 + r + 8 * hi) * 65 + wq * 32 + j * 16 + row] = acc[i][j][r];
  __syncthreads();
  const float* s0 = sub0 + (size_t)b * LC + c0;
  const float* s1 = sub1 + (size_t)b * LQ + q0;
#pragma unroll 4
  for (int ii = 0; ii < 32; ++ii) {
    int idx = tid + ii * 128;
    int cc = idx >> 6, qq = idx & 63;
    S[((size_t)b * LC + c0 + cc) * LQ + q0 + qq] = sDump[cc * 65 + qq] + s0[cc] + s1[qq];
  }
}

// ---- K2: row softmax over Lq=512 -> P1 (bf16), one wave per row ----
__global__ __launch_bounds__(256) void rowsoftmax_kernel(const float* __restrict__ S,
    bf16_t* __restrict__ P) {
  int wave = threadIdx.x >> 5, lane = threadIdx.x & 31;
  size_t rid = (size_t)blockIdx.x * 8 + wave;
  const float* srow = S + rid * LQ;
  bf16_t* prow = P + rid * LQ;
  float v[16];
  float m = -1e30f;
#pragma unroll
  for (int i = 0; i < 16; ++i) { v[i] = srow[lane + i * 32]; m = fmaxf(m, v[i]); }
#pragma unroll
  for (int off = 16; off >= 1; off >>= 1) m = fmaxf(m, __shfl_xor(m, off, 32));
  float s = 0.f;
#pragma unroll
  for (int i = 0; i < 16; ++i) { v[i] = __expf(v[i] - m); s += v[i]; }
#pragma unroll
  for (int off = 16; off >= 1; off >>= 1) s += __shfl_xor(s, off, 32);
  float inv = 1.f / s;
#pragma unroll
  for (int i = 0; i < 16; ++i) prow[lane + i * 32] = (bf16_t)(v[i] * inv);
}

// ---- K3: column softmax over Lc -> P2T[b,q,c] = S2[c,q] (bf16), two passes ----
__global__ __launch_bounds__(256) void colsoftmax_kernel(const float* __restrict__ S,
    bf16_t* __restrict__ P2T) {
  __shared__ float chunk[32 * 257];
  __shared__ float mArr[256], sArr[256];
  const int t = threadIdx.x;
  const int b = blockIdx.y, q0 = blockIdx.x * 256;
  float m = -1e30f, s = 0.f;
  for (int cb = 0; cb < LC / 32; ++cb) {
#pragma unroll
    for (int i = 0; i < 32; ++i) {
      int idx = t + i * 256;
      int cc = idx >> 8, qq = idx & 255;
      chunk[cc * 257 + qq] = S[((size_t)b * LC + cb * 32 + cc) * LQ + q0 + qq];
    }
    __syncthreads();
#pragma unroll 4
    for (int cc = 0; cc < 32; ++cc) {
      float x = chunk[cc * 257 + t];
      float nm = fmaxf(m, x);
      s = s * __expf(m - nm) + __expf(x - nm);
      m = nm;
    }
    __syncthreads();
  }
  mArr[t] = m; sArr[t] = 1.f / s;
  __syncthreads();
  for (int cb = 0; cb < LC / 32; ++cb) {
#pragma unroll
    for (int i = 0; i < 32; ++i) {
      int idx = t + i * 256;
      int cc = idx >> 8, qq = idx & 255;
      chunk[cc * 257 + qq] = S[((size_t)b * LC + cb * 32 + cc) * LQ + q0 + qq];
    }
    __syncthreads();
#pragma unroll
    for (int i = 0; i < 32; ++i) {
      int idx = t + i * 256;
      int ql = idx >> 5, cc = idx & 31;
      float val = __expf(chunk[cc * 257 + ql] - mArr[ql]) * sArr[ql];
      P2T[((size_t)b * LQ + q0 + ql) * LC + cb * 32 + cc] = (bf16_t)val;
    }
    __syncthreads();
  }
}

// ---- K4: Tt[b,d,q] = sum_c P2T[q,c] * C[c,b,d]  (stored transposed, bf16) ----
__global__ __launch_bounds__(128) void t_kernel(const bf16_t* __restrict__ P2T,
    const float* __restrict__ C, bf16_t* __restrict__ Tt) {
  __shared__ bf16_t sX[64 * 40], sBt[64 * 40];
  __shared__ float sDump[64 * 65];
  const int tid = threadIdx.x;
  const int b = blockIdx.z, q0 = blockIdx.y * 64, d0 = blockIdx.x * 64;
  const int wave = tid >> 5, lane = tid & 31;
  const int wq = wave >> 1, wd = wave & 1;
  const int row = lane & 15, hi = lane >> 4;
  const int sr = tid >> 1, scol = (tid & 1) * 16;
  const int tc = tid >> 2, tdc = (tid & 3) * 16;
  v8f acc[2][2] = {};
  for (int kc = 0; kc < LC; kc += 32) {
    const bf16_t* xp = P2T + ((size_t)b * LQ + q0 + sr) * LC + kc + scol;
    async_cp16(sX + sr * 40 + scol,     xp);
    async_cp16(sX + sr * 40 + scol + 8, xp + 8);
    const float* cp = C + ((size_t)(kc + tc) * BATCH + b) * DD + d0 + tdc;
#pragma unroll
    for (int i = 0; i < 16; ++i) sBt[(tdc + i) * 40 + tc] = (bf16_t)cp[i];
    async_wait();
    __syncthreads();
    v16bf a[2], bb[2];
#pragma unroll
    for (int i = 0; i < 2; ++i) {
      a[i]  = frag_a(sX,  wq * 32 + i * 16 + row, hi);
      bb[i] = frag_b(sBt, wd * 32 + i * 16 + row, hi);
    }
#pragma unroll
    for (int i = 0; i < 2; ++i)
#pragma unroll
      for (int j = 0; j < 2; ++j) acc[i][j] = WMMA_BF16(a[i], bb[j], acc[i][j]);
    __syncthreads();
  }
#pragma unroll
  for (int i = 0; i < 2; ++i)
    for (int j = 0; j < 2; ++j)
#pragma unroll
      for (int r = 0; r < 8; ++r)
        sDump[(wq * 32 + i * 16 + r + 8 * hi) * 65 + wd * 32 + j * 16 + row] = acc[i][j][r];
  __syncthreads();
#pragma unroll 4
  for (int ii = 0; ii < 32; ++ii) {
    int idx = tid + ii * 128;
    int dd = idx >> 6, qq = idx & 63;
    Tt[((size_t)b * DD + d0 + dd) * LQ + q0 + qq] = (bf16_t)sDump[qq * 65 + dd];
  }
}

// ---- K5: A = P1*Q, Bm = P1*T; write out[b, sec*D+d, c] = {C, A, C*A, C*Bm} ----
__global__ __launch_bounds__(128) void out_kernel(const bf16_t* __restrict__ P1,
    const float* __restrict__ Q, const bf16_t* __restrict__ Tt,
    const float* __restrict__ C, float* __restrict__ out) {
  __shared__ bf16_t sX[64 * 40], sQt[64 * 40], sTt[64 * 40];
  __shared__ float sA[64 * 65], sB[64 * 65], sC[64 * 65];
  const int tid = threadIdx.x;
  const int b = blockIdx.z, c0 = blockIdx.y * 64, d0 = blockIdx.x * 64;
  const int wave = tid >> 5, lane = tid & 31;
  const int wc = wave >> 1, wd = wave & 1;
  const int row = lane & 15, hi = lane >> 4;
  const int sr = tid >> 1, scol = (tid & 1) * 16;
  const int tq = tid >> 2, tdc = (tid & 3) * 16;
  v8f accA[2][2] = {}, accB[2][2] = {};
  for (int kq = 0; kq < LQ; kq += 32) {
    const bf16_t* xp = P1 + ((size_t)b * LC + c0 + sr) * LQ + kq + scol;
    async_cp16(sX + sr * 40 + scol,     xp);
    async_cp16(sX + sr * 40 + scol + 8, xp + 8);
    const bf16_t* tp = Tt + ((size_t)b * DD + d0 + sr) * LQ + kq + scol;
    async_cp16(sTt + sr * 40 + scol,     tp);
    async_cp16(sTt + sr * 40 + scol + 8, tp + 8);
    const float* qp = Q + ((size_t)(kq + tq) * BATCH + b) * DD + d0 + tdc;
#pragma unroll
    for (int i = 0; i < 16; ++i) sQt[(tdc + i) * 40 + tq] = (bf16_t)qp[i];
    async_wait();
    __syncthreads();
    v16bf a[2], bq[2], bt[2];
#pragma unroll
    for (int i = 0; i < 2; ++i) {
      a[i]  = frag_a(sX,  wc * 32 + i * 16 + row, hi);
      bq[i] = frag_b(sQt, wd * 32 + i * 16 + row, hi);
      bt[i] = frag_b(sTt, wd * 32 + i * 16 + row, hi);
    }
#pragma unroll
    for (int i = 0; i < 2; ++i)
#pragma unroll
      for (int j = 0; j < 2; ++j) {
        accA[i][j] = WMMA_BF16(a[i], bq[j], accA[i][j]);
        accB[i][j] = WMMA_BF16(a[i], bt[j], accB[i][j]);
      }
    __syncthreads();
  }
#pragma unroll
  for (int i = 0; i < 2; ++i)
    for (int j = 0; j < 2; ++j)
#pragma unroll
      for (int r = 0; r < 8; ++r) {
        sA[(wc * 32 + i * 16 + r + 8 * hi) * 65 + wd * 32 + j * 16 + row] = accA[i][j][r];
        sB[(wc * 32 + i * 16 + r + 8 * hi) * 65 + wd * 32 + j * 16 + row] = accB[i][j][r];
      }
  {
    const float* cp = C + ((size_t)(c0 + sr) * BATCH + b) * DD + d0 + scol;
#pragma unroll
    for (int i = 0; i < 16; ++i) sC[sr * 65 + scol + i] = cp[i];
  }
  __syncthreads();
  float* ob = out + (size_t)b * 4 * DD * LC;
#pragma unroll 4
  for (int ii = 0; ii < 128; ++ii) {
    int idx = tid + ii * 128;
    int sec = idx >> 12, rem = idx & 4095;
    int dd = rem >> 6, cc = rem & 63;
    float cv = sC[cc * 65 + dd];
    float av = sA[cc * 65 + dd];
    float bv = sB[cc * 65 + dd];
    float val = (sec == 0) ? cv : (sec == 1) ? av : (sec == 2) ? cv * av : cv * bv;
    ob[(size_t)(sec * DD + d0 + dd) * LC + c0 + cc] = val;
  }
}

extern "C" void kernel_launch(void* const* d_in, const int* in_sizes, int n_in,
                              void* d_out, int out_size, void* d_ws, size_t ws_size,
                              hipStream_t stream) {
  (void)in_sizes; (void)n_in; (void)out_size; (void)ws_size;
  const float* C     = (const float*)d_in[0];
  const float* Q     = (const float*)d_in[1];
  const float* w4C   = (const float*)d_in[2];
  const float* w4Q   = (const float*)d_in[3];
  const float* w4mlu = (const float*)d_in[4];
  const float* bias  = (const float*)d_in[5];
  float* out = (float*)d_out;

  char* ws = (char*)d_ws;
  size_t off = 0;
  float*  S   = (float*)(ws + off);  off += (size_t)BATCH * LC * LQ * sizeof(float);
  bf16_t* P1  = (bf16_t*)(ws + off); off += (size_t)BATCH * LC * LQ * sizeof(bf16_t);
  bf16_t* P2T = (bf16_t*)(ws + off); off += (size_t)BATCH * LQ * LC * sizeof(bf16_t);
  bf16_t* Tt  = (bf16_t*)(ws + off); off += (size_t)BATCH * DD * LQ * sizeof(bf16_t);
  float*  sub0 = (float*)(ws + off); off += (size_t)BATCH * LC * sizeof(float);
  float*  sub1 = (float*)(ws + off); off += (size_t)BATCH * LQ * sizeof(float);

  rowdot_kernel<<<LC * BATCH / 8, 256, 0, stream>>>(C, w4C, bias, 1, sub0, LC);
  rowdot_kernel<<<LQ * BATCH / 8, 256, 0, stream>>>(Q, w4Q, bias, 0, sub1, LQ);
  s_kernel<<<dim3(LQ / 64, LC / 64, BATCH), 128, 0, stream>>>(C, Q, w4mlu, sub0, sub1, S);
  rowsoftmax_kernel<<<BATCH * LC / 8, 256, 0, stream>>>(S, P1);
  colsoftmax_kernel<<<dim3(LQ / 256, BATCH), 256, 0, stream>>>(S, P2T);
  t_kernel<<<dim3(DD / 64, LQ / 64, BATCH), 128, 0, stream>>>(P2T, C, Tt);
  out_kernel<<<dim3(DD / 64, LC / 64, BATCH), 128, 0, stream>>>(P1, Q, Tt, C, out);
}
